// MambaExpert_58067957842023
// MI455X (gfx1250) — matrix operand order repaired and checked by reading, not compiled, not run on hardware
//
#include <hip/hip_runtime.h>

// ---------- sizes ----------
#define D_MODEL 1024
#define D_STATE 16
#define D_CONV  4
#define D_INNER 2048
#define DT_RANK 64
#define BATCH   2
#define SEQ     1024
#define ML      (BATCH * SEQ)            // 2048 rows for all GEMMs
#define XZ_N    (2 * D_INNER)            // 4096
#define DBL_N   (DT_RANK + 2 * D_STATE)  // 96 (logical)
#define DBL_P   128                      // padded to tile width

typedef __attribute__((ext_vector_type(16))) __bf16 v16bf;
typedef __attribute__((ext_vector_type(8)))  __bf16 v8bf;
typedef __attribute__((ext_vector_type(8)))  float  v8f;

// ---------- f32 -> bf16 convert ----------
__global__ void k_f2bf(const float* __restrict__ s, __bf16* __restrict__ d, int n) {
    int i = blockIdx.x * blockDim.x + threadIdx.x;
    if (i < n) d[i] = (__bf16)s[i];
}

// ---------- f32 [K][N] -> bf16 transposed [N][K]; K = 1<<kbits ----------
__global__ void k_f2bf_t(const float* __restrict__ s, __bf16* __restrict__ d,
                         int kbits, int total) {
    int i = blockIdx.x * blockDim.x + threadIdx.x;   // i = n*K + k
    if (i >= total) return;
    int K = 1 << kbits;
    int k = i & (K - 1);
    int n = i >> kbits;
    long long N = (long long)total >> kbits;
    d[i] = (__bf16)s[(size_t)k * (size_t)N + n];
}

// ---------- W_x (2048x96) -> bf16 transposed+padded [128][2048] ----------
__global__ void k_wxt_pack(const float* __restrict__ wx, __bf16* __restrict__ dst) {
    int i = blockIdx.x * blockDim.x + threadIdx.x;   // i = n*2048 + k
    int k = i & (D_INNER - 1);
    int n = i >> 11;
    dst[i] = (n < DBL_N) ? (__bf16)wx[(size_t)k * DBL_N + n] : (__bf16)0.f;
}

// ---------- bf16 WMMA GEMM: C[M,N] f32 = A[M,K] @ B[K,N], B given TRANSPOSED: BT[N][K] ----------
// 256 threads = 8 waves. Block tile 128(M) x 64(N); wave tile 32x32 = 4 WMMAs.
// Double-buffered LDS, one barrier per K-step. M%128==0, N%64==0, K%32==0.
__global__ __launch_bounds__(256) void k_gemm_bf16(
    const __bf16* __restrict__ A, const __bf16* __restrict__ BT, float* __restrict__ C,
    int M, int N, int K) {
    __shared__ __bf16 As[2][128][40];   // [buf][m][k], padded stride (80 B)
    __shared__ __bf16 Bs[2][64][40];    // [buf][n][k]

    const int tid  = threadIdx.x;
    const int lane = tid & 31;
    const int wave = tid >> 5;
    const int wm   = wave & 3;          // 4 M sub-tiles of 32 rows
    const int wn   = wave >> 2;         // 2 N sub-tiles of 32 cols
    const int brow = blockIdx.y * 128;
    const int bcol = blockIdx.x * 64;

    // staging: contiguous 8-bf16 (16 B) chunks
    const int ar0 = tid >> 2;           // A rows 0..63
    const int ar1 = ar0 + 64;           // A rows 64..127
    const int ac  = (tid & 3) * 8;
    const int br  = tid >> 2;           // BT rows (= N cols) 0..63
    const int bc  = (tid & 3) * 8;

    const __bf16* Abase0 = A + (size_t)(brow + ar0) * K + ac;
    const __bf16* Abase1 = A + (size_t)(brow + ar1) * K + ac;
    const __bf16* Bbase  = BT + (size_t)(bcol + br) * K + bc;

    auto stage = [&](int kk, int buf) {
        *(v8bf*)&As[buf][ar0][ac] = *(const v8bf*)(Abase0 + kk);
        *(v8bf*)&As[buf][ar1][ac] = *(const v8bf*)(Abase1 + kk);
        *(v8bf*)&Bs[buf][br][bc]  = *(const v8bf*)(Bbase  + kk);
    };

    v8f acc00 = {}, acc01 = {}, acc10 = {}, acc11 = {};

    const int m0 = wm * 32 + (lane & 15);
    const int m1 = m0 + 16;
    const int n0 = wn * 32 + (lane & 15);
    const int n1 = n0 + 16;
    const int koff  = (lane >> 4) * 8;    // A-frag K sub-offset (ISA 16-bit A layout)
    const int koffb = (lane >> 4) * 8;    // B-frag K sub-offset (same layout as A)

    stage(0, 0);
    __syncthreads();

    for (int k0 = 0; k0 < K; k0 += 32) {
        const int buf = (k0 >> 5) & 1;
        if (k0 + 32 < K) stage(k0 + 32, buf ^ 1);
        if (k0 + 64 < K) {   // keep L2->WGP pipe primed (global_prefetch_b8)
            __builtin_prefetch(Abase0 + k0 + 64, 0, 1);
            __builtin_prefetch(Bbase  + k0 + 64, 0, 1);
        }

        v16bf af0, af1, bf0, bf1;
        {
            v8bf x0 = *(const v8bf*)&As[buf][m0][koff];
            v8bf x1 = *(const v8bf*)&As[buf][m0][koff + 16];
            v8bf y0 = *(const v8bf*)&As[buf][m1][koff];
            v8bf y1 = *(const v8bf*)&As[buf][m1][koff + 16];
            #pragma unroll
            for (int i = 0; i < 8; ++i) {
                af0[i] = x0[i]; af0[8 + i] = x1[i];
                af1[i] = y0[i]; af1[8 + i] = y1[i];
            }
        }
        {
            v8bf x0 = *(const v8bf*)&Bs[buf][n0][koffb];
            v8bf x1 = *(const v8bf*)&Bs[buf][n0][koffb + 16];
            v8bf y0 = *(const v8bf*)&Bs[buf][n1][koffb];
            v8bf y1 = *(const v8bf*)&Bs[buf][n1][koffb + 16];
            #pragma unroll
            for (int i = 0; i < 8; ++i) {
                bf0[i] = x0[i]; bf0[8 + i] = x1[i];
                bf1[i] = y0[i]; bf1[8 + i] = y1[i];
            }
        }

        acc00 = __builtin_amdgcn_wmma_f32_16x16x32_bf16(false, af0, false, bf0, (short)0, acc00, false, false);
        acc01 = __builtin_amdgcn_wmma_f32_16x16x32_bf16(false, af0, false, bf1, (short)0, acc01, false, false);
        acc10 = __builtin_amdgcn_wmma_f32_16x16x32_bf16(false, af1, false, bf0, (short)0, acc10, false, false);
        acc11 = __builtin_amdgcn_wmma_f32_16x16x32_bf16(false, af1, false, bf1, (short)0, acc11, false, false);
        __syncthreads();
    }

    // store: lane l: N = l&15; VGPR r -> M = r + 8*(l>>4)
    const int rbase = brow + wm * 32 + (lane >> 4) * 8;
    const int cbase = bcol + wn * 32 + (lane & 15);
    #pragma unroll
    for (int r = 0; r < 8; ++r) {
        C[(size_t)(rbase +  0 + r) * N + cbase +  0] = acc00[r];
        C[(size_t)(rbase +  0 + r) * N + cbase + 16] = acc01[r];
        C[(size_t)(rbase + 16 + r) * N + cbase +  0] = acc10[r];
        C[(size_t)(rbase + 16 + r) * N + cbase + 16] = acc11[r];
    }
}

// ---------- causal depthwise conv (width 4) + bias + SiLU; writes f32 and bf16 ----------
__global__ void k_conv_silu(const float* __restrict__ xz, const float* __restrict__ conv_w,
                            const float* __restrict__ conv_b,
                            float* __restrict__ xc, __bf16* __restrict__ xcbf) {
    int idx = blockIdx.x * blockDim.x + threadIdx.x;   // (b*SEQ+t)*D_INNER + d
    int d  = idx & (D_INNER - 1);
    int bt = idx >> 11;
    int t  = bt & (SEQ - 1);
    int b  = bt >> 10;
    float acc = conv_b[d];
    #pragma unroll
    for (int j = 0; j < D_CONV; ++j) {
        int ts = t - (D_CONV - 1) + j;
        if (ts >= 0)
            acc += conv_w[d * D_CONV + j] * xz[((size_t)(b * SEQ + ts)) * XZ_N + d];
    }
    float s = acc / (1.f + __expf(-acc));
    xc[idx]   = s;
    xcbf[idx] = (__bf16)s;
}

// ---------- pack dt-rank slice of padded dbl to bf16 [ML, DT_RANK] ----------
__global__ void k_dtr_pack(const float* __restrict__ dbl, __bf16* __restrict__ dtrbf) {
    int i = blockIdx.x * blockDim.x + threadIdx.x;     // 0 .. ML*DT_RANK-1
    int r  = i & (DT_RANK - 1);
    int bt = i >> 6;
    dtrbf[i] = (__bf16)dbl[(size_t)bt * DBL_P + r];
}

// ---------- dt = softplus(dt_lin + b_dt) in place ----------
__global__ void k_dt_act(float* __restrict__ dt, const float* __restrict__ b_dt) {
    int i = blockIdx.x * blockDim.x + threadIdx.x;
    int d = i & (D_INNER - 1);
    float v = dt[i] + b_dt[d];
    dt[i] = (v > 20.f) ? v : log1pf(__expf(v));
}

// ---------- sequential selective scan, fused D-skip + SiLU(z) gate, bf16 output ----------
__global__ __launch_bounds__(256) void k_scan(
    const float* __restrict__ dt, const float* __restrict__ dbl,
    const float* __restrict__ xc, const float* __restrict__ xz,
    const float* __restrict__ A_log, const float* __restrict__ D_skip,
    __bf16* __restrict__ ybf) {
    const int d = blockIdx.x * blockDim.x + threadIdx.x;   // channel 0..2047
    const int b = blockIdx.y;
    float A[D_STATE], h[D_STATE];
    #pragma unroll
    for (int s = 0; s < D_STATE; ++s) {
        A[s] = -__expf(A_log[d * D_STATE + s]);
        h[s] = 0.f;
    }
    const float Dsk = D_skip[d];
    const size_t base = (size_t)b * SEQ;
    for (int t = 0; t < SEQ; ++t) {
        const size_t bt = base + t;
        const float dtv = dt[bt * D_INNER + d];
        const float xv  = xc[bt * D_INNER + d];
        const float zv  = xz[bt * XZ_N + D_INNER + d];
        const float4* Bp = (const float4*)(dbl + bt * DBL_P + DT_RANK);
        const float4* Cp = (const float4*)(dbl + bt * DBL_P + DT_RANK + D_STATE);
        const float dx = dtv * xv;
        float y = 0.f;
        #pragma unroll
        for (int q = 0; q < 4; ++q) {
            float4 Bv = Bp[q];
            float4 Cv = Cp[q];
            float* hq = &h[4 * q];
            hq[0] = __expf(dtv * A[4 * q + 0]) * hq[0] + dx * Bv.x; y += hq[0] * Cv.x;
            hq[1] = __expf(dtv * A[4 * q + 1]) * hq[1] + dx * Bv.y; y += hq[1] * Cv.y;
            hq[2] = __expf(dtv * A[4 * q + 2]) * hq[2] + dx * Bv.z; y += hq[2] * Cv.z;
            hq[3] = __expf(dtv * A[4 * q + 3]) * hq[3] + dx * Bv.w; y += hq[3] * Cv.w;
        }
        const float out  = y + xv * Dsk;
        const float gate = zv / (1.f + __expf(-zv));
        ybf[bt * D_INNER + d] = (__bf16)(out * gate);
    }
}

// ---------- host ----------
extern "C" void kernel_launch(void* const* d_in, const int* in_sizes, int n_in,
                              void* d_out, int out_size, void* d_ws, size_t ws_size,
                              hipStream_t stream) {
    const float* x      = (const float*)d_in[0];
    const float* W_in   = (const float*)d_in[1];
    const float* conv_w = (const float*)d_in[2];
    const float* conv_b = (const float*)d_in[3];
    const float* W_x    = (const float*)d_in[4];
    const float* W_dt   = (const float*)d_in[5];
    const float* b_dt   = (const float*)d_in[6];
    const float* A_log  = (const float*)d_in[7];
    const float* D_skip = (const float*)d_in[8];
    const float* W_out  = (const float*)d_in[9];
    float* out = (float*)d_out;

    // workspace carving (256-B aligned)
    size_t off = 0;
    auto carve = [&](size_t bytes) -> void* {
        void* p = (char*)d_ws + off;
        off += (bytes + 255) & ~(size_t)255;
        return p;
    };
    __bf16* xbf    = (__bf16*)carve((size_t)ML * D_MODEL * 2);        // 4 MB
    __bf16* WinT   = (__bf16*)carve((size_t)D_MODEL * XZ_N * 2);      // 8 MB  [4096][1024]
    float*  xzf    = (float*) carve((size_t)ML * XZ_N * 4);           // 32 MB
    float*  xcf    = (float*) carve((size_t)ML * D_INNER * 4);        // 16 MB
    __bf16* xcbf   = (__bf16*)carve((size_t)ML * D_INNER * 2);        // 8 MB
    __bf16* WxT    = (__bf16*)carve((size_t)DBL_P * D_INNER * 2);     // [128][2048]
    float*  dblf   = (float*) carve((size_t)ML * DBL_P * 4);          // padded
    __bf16* dtrbf  = (__bf16*)carve((size_t)ML * DT_RANK * 2);
    __bf16* WdtT   = (__bf16*)carve((size_t)D_INNER * DT_RANK * 2);   // [2048][64]
    float*  dtf    = (float*) carve((size_t)ML * D_INNER * 4);        // 16 MB
    __bf16* ybf    = (__bf16*)carve((size_t)ML * D_INNER * 2);        // 8 MB
    __bf16* WoutT  = (__bf16*)carve((size_t)D_MODEL * D_INNER * 2);   // [1024][2048]

    const int T = 256;
    auto g1 = [&](int n) { return dim3((unsigned)((n + T - 1) / T)); };

    // 1. convert x; convert+transpose weights to BT[N][K] bf16 (W_x also padded)
    k_f2bf<<<g1(ML * D_MODEL), T, 0, stream>>>(x, xbf, ML * D_MODEL);
    k_f2bf_t<<<g1(XZ_N * D_MODEL),    T, 0, stream>>>(W_in,  WinT,  10, XZ_N * D_MODEL);    // K=1024
    k_wxt_pack<<<g1(DBL_P * D_INNER), T, 0, stream>>>(W_x,   WxT);                          // K=2048 padded
    k_f2bf_t<<<g1(D_INNER * DT_RANK), T, 0, stream>>>(W_dt,  WdtT,  6,  D_INNER * DT_RANK); // K=64
    k_f2bf_t<<<g1(D_MODEL * D_INNER), T, 0, stream>>>(W_out, WoutT, 11, D_MODEL * D_INNER); // K=2048

    // 2. xz = x @ W_in   [2048 x 1024 x 4096]
    k_gemm_bf16<<<dim3(XZ_N / 64, ML / 128), T, 0, stream>>>(xbf, WinT, xzf, ML, XZ_N, D_MODEL);

    // 3. causal depthwise conv + SiLU
    k_conv_silu<<<g1(ML * D_INNER), T, 0, stream>>>(xzf, conv_w, conv_b, xcf, xcbf);

    // 4. dbl = xc @ W_x_pad  [2048 x 2048 x 128]
    k_gemm_bf16<<<dim3(DBL_P / 64, ML / 128), T, 0, stream>>>(xcbf, WxT, dblf, ML, DBL_P, D_INNER);

    // 5. pack dt-rank slice; 6. dt_lin = dtr @ W_dt  [2048 x 64 x 2048]; 7. softplus
    k_dtr_pack<<<g1(ML * DT_RANK), T, 0, stream>>>(dblf, dtrbf);
    k_gemm_bf16<<<dim3(D_INNER / 64, ML / 128), T, 0, stream>>>(dtrbf, WdtT, dtf, ML, D_INNER, DT_RANK);
    k_dt_act<<<g1(ML * D_INNER), T, 0, stream>>>(dtf, b_dt);

    // 8. sequential scan + D-skip + SiLU(z) gate -> ybf
    k_scan<<<dim3(D_INNER / T, BATCH), T, 0, stream>>>(dtf, dblf, xcf, xzf, A_log, D_skip, ybf);

    // 9. out = y @ W_out  [2048 x 2048 x 1024] -> d_out (f32)
    k_gemm_bf16<<<dim3(D_MODEL / 64, ML / 128), T, 0, stream>>>(ybf, WoutT, out, ML, D_MODEL, D_INNER);
}
